// KeyValueMemoryNetwork_80255758893374
// MI455X (gfx1250) — matrix operand order repaired and checked by reading, not compiled, not run on hardware
//
#include <hip/hip_runtime.h>
#include <stdint.h>

// ---------------------------------------------------------------------------
// KeyValueMemoryNetwork forward for MI455X (gfx1250, wave32, WMMA bf16)
//   q = (hidden @ weight) / sqrt(E)        [B,S,E]   -> bf16 (scale folded in)
//   u = q @ k^T ;  delta = exp(u) * clip(mask,0,1)
//   o = (delta @ v) / (rowsum(delta)+1e-10)
// Single pass over t (reference has no max-subtraction -> no rescale needed).
// Attention block: 32 s-rows, 8 waves (i = s-subtile 0/1, j = N-tile/e-chunk),
// ping-pong delta buffers -> one barrier per 64-wide t-step, mask prefetched.
// ---------------------------------------------------------------------------

#define BB 8
#define SS 2048
#define EE 512

typedef __attribute__((ext_vector_type(16))) __bf16 v16bf;
typedef __attribute__((ext_vector_type(8)))  __bf16 v8bf;
typedef __attribute__((ext_vector_type(8)))  float  v8f;

// ---------------------------- prep kernels ---------------------------------

__global__ void prep_k_kernel(const int* __restrict__ key_seq,
                              const float* __restrict__ key_emb,
                              __bf16* __restrict__ k_bf) {
  int64_t idx = (int64_t)blockIdx.x * blockDim.x + threadIdx.x; // over B*S*E
  int e = (int)(idx % EE);
  int64_t bt = idx / EE;
  int row = key_seq[bt];
  k_bf[idx] = (__bf16)key_emb[(int64_t)row * EE + e];
}

__global__ void prep_vT_kernel(const int* __restrict__ value_seq,
                               const float* __restrict__ value_emb,
                               __bf16* __restrict__ vT_bf) {
  int64_t idx = (int64_t)blockIdx.x * blockDim.x + threadIdx.x; // over B*E*S
  int t = (int)(idx % SS);
  int64_t be = idx / SS;
  int e = (int)(be % EE);
  int b = (int)(be / EE);
  int row = value_seq[(int64_t)b * SS + t];
  vT_bf[idx] = (__bf16)value_emb[(int64_t)row * EE + e];
}

__global__ void prep_wT_kernel(const float* __restrict__ weight,
                               __bf16* __restrict__ wT_bf) {
  int idx = blockIdx.x * blockDim.x + threadIdx.x; // over E*E, layout [n][k]
  int k = idx % EE;
  int n = idx / EE;
  wT_bf[idx] = (__bf16)weight[(int64_t)k * EE + n];
}

// ---------------------------- projection -----------------------------------
// q_bf[m][e] = bf16( (hidden @ weight)[m][e] * 1/sqrt(E) ),  m = b*S+s
__global__ __launch_bounds__(128) void proj_q_kernel(
    const float* __restrict__ hidden, const __bf16* __restrict__ wT,
    __bf16* __restrict__ q_bf) {
  const float inv_scale = 0.04419417382415922f; // 1/sqrt(512)
  int lane = threadIdx.x & 31;
  int wave = threadIdx.x >> 5;
  int half = lane >> 4;
  int ln   = lane & 15;
  int64_t row0 = (int64_t)blockIdx.y * 16;
  int e0 = blockIdx.x * 256 + wave * 64;

  v8f acc[4];
#pragma unroll
  for (int n = 0; n < 4; ++n) acc[n] = (v8f)(0.0f);

  for (int kc = 0; kc < EE / 32; ++kc) {
    int kb = kc * 32;
    const float* arow = hidden + (row0 + ln) * EE + kb + half * 16;
    v16bf a;
#pragma unroll
    for (int i = 0; i < 8; ++i) {
      a[i]     = (__bf16)arow[i];
      a[i + 8] = (__bf16)arow[8 + i];
    }
#pragma unroll
    for (int n = 0; n < 4; ++n) {
      const v16bf bfrag =
          *(const v16bf*)(wT + (int64_t)(e0 + n * 16 + ln) * EE + kb + half * 16);
      acc[n] = __builtin_amdgcn_wmma_f32_16x16x32_bf16(
          false, a, false, bfrag, (short)0, acc[n], false, false);
    }
  }
#pragma unroll
  for (int n = 0; n < 4; ++n)
#pragma unroll
    for (int r = 0; r < 8; ++r)
      q_bf[(row0 + r + 8 * half) * EE + e0 + n * 16 + ln] =
          (__bf16)(acc[n][r] * inv_scale);
}

// ---------------------------- attention ------------------------------------
// grid = (S/32, B), block = 256 (8 waves). Block owns 32 s-rows.
// wave w: i = w>>2 (s-subtile of 16 rows), j = w&3 (u N-tile / 128-wide e-chunk)
__global__ __launch_bounds__(256) void attn_kernel(
    const __bf16* __restrict__ q_bf, const __bf16* __restrict__ k_bf,
    const __bf16* __restrict__ vT_bf, const float* __restrict__ mask,
    float* __restrict__ out) {
  __shared__ __bf16 q_lds[32 * EE];            // 32 KB: q tile [m][e]
  __shared__ __bf16 delta_lds[2][2][16 * 64];  // 8 KB : [pingpong][subtile][m][t]
  __shared__ float  rowsum_lds[32];

  int tid  = threadIdx.x;
  int lane = tid & 31;
  int wave = tid >> 5;
  int half = lane >> 4;
  int ln   = lane & 15;
  int sub  = wave >> 2;  // i: s-subtile 0/1
  int j    = wave & 3;   // N-tile (u phase) / e-chunk (PV phase)
  int b  = blockIdx.y;
  int s0 = blockIdx.x * 32;

  // cooperative load of the q tile (32*512 bf16 = 2048 x 16B chunks)
  {
    const uint4* src = (const uint4*)(q_bf + ((int64_t)b * SS + s0) * EE);
    uint4* dst = (uint4*)q_lds;
#pragma unroll
    for (int i = 0; i < 8; ++i) dst[tid + i * 256] = src[tid + i * 256];
  }
  __syncthreads();

  v8f acc[8];
#pragma unroll
  for (int n = 0; n < 8; ++n) acc[n] = (v8f)(0.0f);
  float rs[8];
#pragma unroll
  for (int r = 0; r < 8; ++r) rs[r] = 0.0f;

  int e_base = j * 128;
  const __bf16* qrow_lo = q_lds + (sub * 16 + ln) * EE;

  for (int t0 = 0; t0 < SS; t0 += 64) {
    int p  = (t0 >> 6) & 1;  // ping-pong buffer parity
    int tN = t0 + j * 16;

    // ---- phase 1: u N-tile j for subtile `sub`, columns [tN, tN+16) ----
    v8f u = (v8f)(0.0f);
    for (int kc = 0; kc < EE / 32; ++kc) {
      int kb = kc * 32;
      v8bf lo = *(const v8bf*)(qrow_lo + kb + half * 16);
      v8bf hi = *(const v8bf*)(qrow_lo + kb + half * 16 + 8);
      v16bf a;
#pragma unroll
      for (int i = 0; i < 8; ++i) { a[i] = lo[i]; a[i + 8] = hi[i]; }
      const v16bf bfrag =
          *(const v16bf*)(k_bf + ((int64_t)b * SS + tN + ln) * EE + kb + half * 16);
      u = __builtin_amdgcn_wmma_f32_16x16x32_bf16(
          false, a, false, bfrag, (short)0, u, false, false);
    }

    // exp * clip(mask), rowsum partials, delta -> LDS (bf16); prefetch next mask
    const float* mrow = mask + ((int64_t)b * SS + s0 + sub * 16) * SS + tN + ln;
#pragma unroll
    for (int r = 0; r < 8; ++r) {
      int M = r + 8 * half;
      float mv = mrow[(int64_t)M * SS];
      if (t0 + 64 < SS)  // stream-prefetch next t-step of the only HBM operand
        __builtin_prefetch(mrow + (int64_t)M * SS + 64, 0, 1);
      mv = fminf(fmaxf(mv, 0.0f), 1.0f);
      float d = __expf(u[r]) * mv;
      rs[r] += d;
      delta_lds[p][sub][M * 64 + j * 16 + ln] = (__bf16)d;
    }
    __syncthreads();  // single barrier per t-step (ping-pong on delta_lds)

    // ---- phase 2: o += delta(16x64) @ v(64 x 128-wide e-chunk j) ----
    const __bf16* drow = delta_lds[p][sub] + ln * 64;
#pragma unroll
    for (int kf = 0; kf < 2; ++kf) {
      v8bf lo = *(const v8bf*)(drow + kf * 32 + half * 16);
      v8bf hi = *(const v8bf*)(drow + kf * 32 + half * 16 + 8);
      v16bf a;
#pragma unroll
      for (int i = 0; i < 8; ++i) { a[i] = lo[i]; a[i + 8] = hi[i]; }
#pragma unroll
      for (int n = 0; n < 8; ++n) {
        const v16bf bfrag = *(const v16bf*)(
            vT_bf + ((int64_t)b * EE + e_base + n * 16 + ln) * SS +
            t0 + kf * 32 + half * 16);
        acc[n] = __builtin_amdgcn_wmma_f32_16x16x32_bf16(
            false, a, false, bfrag, (short)0, acc[n], false, false);
      }
    }
  }

  // ---- rowsum reduction (epsilon folded into init) ----
  __syncthreads();
  if (tid < 32) rowsum_lds[tid] = 1e-10f;
  __syncthreads();
#pragma unroll
  for (int r = 0; r < 8; ++r)
    atomicAdd(&rowsum_lds[sub * 16 + r + 8 * half], rs[r]);
  __syncthreads();

  // ---- normalize + store ----
  float* orow = out + ((int64_t)b * SS + s0 + sub * 16) * EE;
#pragma unroll
  for (int n = 0; n < 8; ++n)
#pragma unroll
    for (int r = 0; r < 8; ++r) {
      int M = r + 8 * half;
      orow[(int64_t)M * EE + e_base + n * 16 + ln] =
          acc[n][r] / rowsum_lds[sub * 16 + M];
    }
}

// ---------------------------- launcher -------------------------------------
extern "C" void kernel_launch(void* const* d_in, const int* in_sizes, int n_in,
                              void* d_out, int out_size, void* d_ws,
                              size_t ws_size, hipStream_t stream) {
  (void)in_sizes; (void)n_in; (void)out_size; (void)ws_size;
  const int*   key_seq   = (const int*)d_in[0];
  const int*   value_seq = (const int*)d_in[1];
  const float* hidden    = (const float*)d_in[2];
  const float* mask      = (const float*)d_in[3];
  // d_in[4] nan_matrix: unused by forward
  const float* key_emb   = (const float*)d_in[5];
  const float* value_emb = (const float*)d_in[6];
  const float* weight    = (const float*)d_in[7];
  float* out = (float*)d_out;

  // workspace layout (bf16):  q | k | vT | wT   = 16+16+16+0.5 MiB
  const size_t SZ = (size_t)BB * SS * EE * sizeof(__bf16); // 16 MiB
  char* ws = (char*)d_ws;
  __bf16* q_bf  = (__bf16*)(ws);
  __bf16* k_bf  = (__bf16*)(ws + SZ);
  __bf16* vT_bf = (__bf16*)(ws + 2 * SZ);
  __bf16* wT_bf = (__bf16*)(ws + 3 * SZ);

  int threads = 256;
  int64_t nKV = (int64_t)BB * SS * EE;
  prep_k_kernel <<<(int)(nKV / threads), threads, 0, stream>>>(key_seq, key_emb, k_bf);
  prep_vT_kernel<<<(int)(nKV / threads), threads, 0, stream>>>(value_seq, value_emb, vT_bf);
  prep_wT_kernel<<<(EE * EE) / threads, threads, 0, stream>>>(weight, wT_bf);

  dim3 gproj(EE / 256, (BB * SS) / 16);
  proj_q_kernel<<<gproj, 128, 0, stream>>>(hidden, wT_bf, q_bf);

  dim3 gattn(SS / 32, BB);
  attn_kernel<<<gattn, 256, 0, stream>>>(q_bf, k_bf, vT_bf, mask, out);
}
